// EnhancedTransformer_64046552318166
// MI455X (gfx1250) — matrix operand order repaired and checked by reading, not compiled, not run on hardware
//
#include <hip/hip_runtime.h>
#include <hip/hip_bf16.h>
#include <math.h>

// ---------------------------------------------------------------------------
// EnhancedTransformer forward for MI455X (gfx1250, wave32, WMMA).
// Memory-bound: ~1GB fp32 weights streamed once (~45us floor @ 23.3 TB/s);
// all matmuls via v_wmma_f32_16x16x32_bf16 (native bf16 convert, f32
// accumulate). GEMM: double-buffered LDS + prefetch. Attention: WMMA flash
// attention with streaming softmax, 4 independent waves per block.
// ---------------------------------------------------------------------------

#define D_MODEL 1024
#define NHEAD   16
#define DHEAD   64
#define FFDIM   4096
#define B_SZ    2
#define S_SRC_L 1024
#define S_TGT_L 512
#define NENC    6
#define NDEC    6
#define VTGT    32000
#define LN_EPS  1e-5f
#define NEGINF  (-1e9f)

typedef __attribute__((ext_vector_type(16))) __bf16        v16bf;
typedef __attribute__((ext_vector_type(8)))  __bf16        v8bf;
typedef __attribute__((ext_vector_type(2)))  __bf16        v2bf;
typedef __attribute__((ext_vector_type(8)))  float         v8f;

// f32 -> bf16 (RTNE fptrunc; backend selects v_cvt_pk_bf16_f32 on gfx1250).
__device__ __forceinline__ unsigned short f2bf_u(float f) {
  __bf16 h = (__bf16)f;
  return __builtin_bit_cast(unsigned short, h);
}
// Pack two f32 into one dword of bf16x2 == one v_cvt_pk_bf16_f32.
__device__ __forceinline__ unsigned int pack2bf(float lo, float hi) {
  v2bf v;
  v[0] = (__bf16)lo;
  v[1] = (__bf16)hi;
  return __builtin_bit_cast(unsigned int, v);
}

// ---------------------------------------------------------------------------
// GEMM: C[M,N] = resid + alpha * (A[M,K] @ W[K,N] + bias[N])
// alpha = cscale * (*alpha_ptr or 1).  M % 128 == 0, K % 32 == 0, any N.
// Block: 256 thr = 8 waves as 4(M)x2(N); wave tile 32x64 = 2x4 WMMA tiles.
// Double-buffered LDS (bf16, packed u32 stores), k+2 prefetch.
// ---------------------------------------------------------------------------
__global__ __launch_bounds__(256)
void gemm_bf16w(const float* __restrict__ A, const float* __restrict__ W,
                const float* __restrict__ bias, const float* __restrict__ resid,
                const float* __restrict__ alpha_ptr, float cscale,
                float* __restrict__ C, int M, int N, int K)
{
  __shared__ unsigned int Alds[2][128][16];   // [buf][m][k/2] packed bf16x2
  __shared__ unsigned int Wlds[2][128][16];   // [buf][n][k/2] (W transposed)
  const int tid  = threadIdx.x;
  const int l    = tid & 31;
  const int w    = tid >> 5;
  const int half = (l >> 4) & 1;
  const int ln   = l & 15;
  const int wm   = w & 3;                     // M offset wm*32
  const int wn   = w >> 2;                    // N offset wn*64

  v8f acc[2][4];
  for (int mt = 0; mt < 2; ++mt)
    for (int nt = 0; nt < 4; ++nt)
      for (int j = 0; j < 8; ++j) acc[mt][nt][j] = 0.0f;

  const int row_ld = tid >> 1;                // 0..127
  const int kc     = (tid & 1) * 16;
  const size_t grow = (size_t)blockIdx.y * 128 + row_ld;
  const int    gn   = blockIdx.x * 128 + row_ld;
  const float* aBase = A + grow * (size_t)K + kc;
  const bool   wok   = (gn < N);

  float ar[16], wr[16];
  auto loadTiles = [&](int kt) {
    const float* ap = aBase + kt;
    for (int i = 0; i < 16; ++i) ar[i] = ap[i];
    for (int i = 0; i < 16; ++i)
      wr[i] = wok ? W[(size_t)(kt + kc + i) * N + gn] : 0.0f;
  };
  auto stage = [&](int buf) {
    for (int i = 0; i < 16; i += 2) {
      Alds[buf][row_ld][(kc + i) >> 1] = pack2bf(ar[i], ar[i + 1]);
      Wlds[buf][row_ld][(kc + i) >> 1] = pack2bf(wr[i], wr[i + 1]);
    }
  };

  loadTiles(0);
  stage(0);
  __syncthreads();

  int cur = 0;
  const int abase = half ? 16 : 0;            // A-frag K base within tile
  for (int kt = 0; kt < K; kt += 32) {
    const bool more = (kt + 32) < K;
    if (more) {
      if (kt + 64 < K) {                      // warm L2/WGP$ for k+2 tile
        __builtin_prefetch(aBase + kt + 64, 0, 0);
        if (wok) __builtin_prefetch(&W[(size_t)(kt + 64 + kc) * N + gn], 0, 0);
      }
      loadTiles(kt + 32);
    }

    const unsigned short* Ab = (const unsigned short*)Alds[cur];
    const unsigned short* Wb = (const unsigned short*)Wlds[cur];
    v16bf af[2];
    for (int mt = 0; mt < 2; ++mt) {
      int r = wm * 32 + mt * 16 + ln;
      v8bf lo = *(const v8bf*)(Ab + r * 32 + abase);
      v8bf hi = *(const v8bf*)(Ab + r * 32 + abase + 8);
      v16bf a;
      for (int e = 0; e < 8; ++e) { a[e] = lo[e]; a[e + 8] = hi[e]; }
      af[mt] = a;
    }
    for (int nt = 0; nt < 4; ++nt) {
      int n = wn * 64 + nt * 16 + ln;
      v16bf bm = *(const v16bf*)(Wb + n * 32 + half * 16);
      acc[0][nt] = __builtin_amdgcn_wmma_f32_16x16x32_bf16(
          false, af[0], false, bm, (short)0, acc[0][nt], false, false);
      acc[1][nt] = __builtin_amdgcn_wmma_f32_16x16x32_bf16(
          false, af[1], false, bm, (short)0, acc[1][nt], false, false);
    }

    if (more) stage(cur ^ 1);
    __syncthreads();
    cur ^= 1;
  }

  const float al = cscale * (alpha_ptr ? alpha_ptr[0] : 1.0f);
  const int row0 = blockIdx.y * 128 + wm * 32;
  const int col0 = blockIdx.x * 128 + wn * 64;
  for (int nt = 0; nt < 4; ++nt) {
    int c = col0 + nt * 16 + ln;
    if (c >= N) continue;
    float bv = bias ? bias[c] : 0.0f;
    for (int mt = 0; mt < 2; ++mt)
      for (int j = 0; j < 8; ++j) {
        int r = row0 + mt * 16 + j + half * 8;
        size_t idx = (size_t)r * N + c;
        float val = al * (acc[mt][nt][j] + bv);
        if (resid) val += resid[idx];
        C[idx] = val;
      }
  }
}

// ---------------------------------------------------------------------------
// MQA flash attention. Q:[B*T, D] (col h*64+d), K/V:[B*S, 64], O:[B*T, D].
// 128 threads = 4 independent waves; wave handles one 16-row Q tile of (b,h).
// No block barriers (causal early-exit diverges per wave): the per-wave LDS
// prob-transpose is ordered by hardware (per-wave DS in-order) + explicit
// s_wait_dscnt compiler fences.
// ---------------------------------------------------------------------------
__global__ __launch_bounds__(128)
void mqa_flash(const float* __restrict__ q, const float* __restrict__ k,
               const float* __restrict__ v, float* __restrict__ o,
               int T, int S, int causal)
{
  __shared__ unsigned short P[4][16][32];     // per-wave probs bounce (bf16)
  const int tid  = threadIdx.x;
  const int wv   = tid >> 5;
  const int l    = tid & 31;
  const int half = l >> 4, ln = l & 15;
  const int tT4  = T >> 6;                    // 64-row tiles per sequence
  const int bid  = blockIdx.x;
  const int t0   = ((bid % tT4) * 4 + wv) << 4;
  const int h    = (bid / tT4) % NHEAD;
  const int b    = bid / (tT4 * NHEAD);
  unsigned short (*Pw)[32] = P[wv];

  // Q A-fragments (16x64 -> 2 x 16x32)
  v16bf aq[2];
  {
    const float* qp = q + (size_t)(b * T + t0 + ln) * D_MODEL + h * DHEAD;
    for (int f = 0; f < 2; ++f) {
      v16bf t;
      int base = f * 32 + half * 16;
      for (int e = 0; e < 8; ++e) t[e]     = (__bf16)qp[base + e];
      for (int e = 0; e < 8; ++e) t[8 + e] = (__bf16)qp[base + 8 + e];
      aq[f] = t;
    }
  }

  float m[8], ls[8];
  v8f oacc[4];
  for (int j = 0; j < 8; ++j) { m[j] = -3e38f; ls[j] = 0.0f; }
  for (int nt = 0; nt < 4; ++nt)
    for (int j = 0; j < 8; ++j) oacc[nt][j] = 0.0f;

  const int send = causal ? (t0 + 16) : S;
  for (int s0 = 0; s0 < send; s0 += 32) {
    // scores: two 16x16 tiles (cols s0.. and s0+16..)
    v8f ct[2];
    for (int p = 0; p < 2; ++p) {
      v8f c;
      for (int j = 0; j < 8; ++j) c[j] = 0.0f;
      const float* kp = k + (size_t)(b * S + s0 + p * 16 + ln) * DHEAD + half * 16;
      for (int f = 0; f < 2; ++f) {
        v16bf t;
        for (int e = 0; e < 16; ++e) t[e] = (__bf16)kp[f * 32 + e];
        c = __builtin_amdgcn_wmma_f32_16x16x32_bf16(
            false, aq[f], false, t, (short)0, c, false, false);
      }
      if (causal) {
        int cs = s0 + p * 16 + ln;
        for (int j = 0; j < 8; ++j)
          if (cs > t0 + j + half * 8) c[j] = NEGINF;
      }
      ct[p] = c;
    }
    // streaming softmax (rows live one-per-lane-half in C layout)
    float p0[8], p1[8];
    for (int j = 0; j < 8; ++j) {
      float r = fmaxf(ct[0][j], ct[1][j]);
      for (int mk = 1; mk < 16; mk <<= 1) r = fmaxf(r, __shfl_xor(r, mk, 32));
      float mn  = fmaxf(m[j], r);
      float fac = __expf(m[j] - mn);
      p0[j] = __expf(ct[0][j] - mn);
      p1[j] = __expf(ct[1][j] - mn);
      float rs = p0[j] + p1[j];
      for (int mk = 1; mk < 16; mk <<= 1) rs += __shfl_xor(rs, mk, 32);
      ls[j] = ls[j] * fac + rs;
      m[j] = mn;
      for (int nt = 0; nt < 4; ++nt) oacc[nt][j] *= fac;
    }
    // transpose probs C-layout -> A-layout through this wave's LDS slice.
    asm volatile("s_wait_dscnt 0x0" ::: "memory");   // prior reads done
    for (int j = 0; j < 8; ++j) {
      Pw[j + half * 8][ln]      = f2bf_u(p0[j]);
      Pw[j + half * 8][16 + ln] = f2bf_u(p1[j]);
    }
    asm volatile("s_wait_dscnt 0x0" ::: "memory");   // stores visible (per-wave in-order)
    v16bf apv;
    {
      v8bf lo = *(const v8bf*)&Pw[ln][half * 16];
      v8bf hi = *(const v8bf*)&Pw[ln][half * 16 + 8];
      for (int e = 0; e < 8; ++e) { apv[e] = lo[e]; apv[e + 8] = hi[e]; }
    }
    // O += P(16x32) @ V(32x64)
    for (int nt = 0; nt < 4; ++nt) {
      v16bf t;
      const float* vp = v + (size_t)(b * S + s0 + half * 16) * DHEAD + nt * 16 + ln;
      for (int e = 0; e < 16; ++e) t[e] = (__bf16)vp[(size_t)e * DHEAD];
      oacc[nt] = __builtin_amdgcn_wmma_f32_16x16x32_bf16(
          false, apv, false, t, (short)0, oacc[nt], false, false);
    }
  }
  for (int nt = 0; nt < 4; ++nt)
    for (int j = 0; j < 8; ++j) {
      int row = t0 + j + half * 8;
      int col = h * DHEAD + nt * 16 + ln;
      o[(size_t)(b * T + row) * D_MODEL + col] = oacc[nt][j] / ls[j];
    }
}

// ---------------------------------------------------------------------------
// LayerNorm: one block per row of D=1024.
// ---------------------------------------------------------------------------
__global__ __launch_bounds__(256)
void ln_kernel(const float* __restrict__ x, const float* __restrict__ g,
               const float* __restrict__ b, float* __restrict__ y, int Dm)
{
  __shared__ float red[256];
  const int row = blockIdx.x, tid = threadIdx.x;
  const float* xr = x + (size_t)row * Dm;
  float s = 0.0f;
  for (int d = tid; d < Dm; d += 256) s += xr[d];
  red[tid] = s; __syncthreads();
  for (int o = 128; o > 0; o >>= 1) { if (tid < o) red[tid] += red[tid + o]; __syncthreads(); }
  const float mean = red[0] / Dm;
  __syncthreads();
  float vs = 0.0f;
  for (int d = tid; d < Dm; d += 256) { float t = xr[d] - mean; vs += t * t; }
  red[tid] = vs; __syncthreads();
  for (int o = 128; o > 0; o >>= 1) { if (tid < o) red[tid] += red[tid + o]; __syncthreads(); }
  const float rstd = rsqrtf(red[0] / Dm + LN_EPS);
  float* yr = y + (size_t)row * Dm;
  for (int d = tid; d < Dm; d += 256) yr[d] = (xr[d] - mean) * rstd * g[d] + b[d];
}

// ---------------------------------------------------------------------------
// Embedding lookup * sqrt(D) + RoPE (full-D rotation, half = D/2).
// ---------------------------------------------------------------------------
__global__ void embed_rope(const int* __restrict__ toks, const float* __restrict__ emb,
                           float* __restrict__ out, int S)
{
  const int row = blockIdx.x;
  const int pos = row % S;
  const int tok = toks[row];
  const float* er = emb + (size_t)tok * D_MODEL;
  float* orow = out + (size_t)row * D_MODEL;
  const float cexp = 2.0f * 9.210340371976184f / (float)D_MODEL;  // ln(1e4)*2/D
  for (int d = threadIdx.x; d < D_MODEL / 2; d += blockDim.x) {
    float invf = __expf(-cexp * (float)d);
    float ang = (float)pos * invf;
    float cs = __cosf(ang), sn = __sinf(ang);
    float e1 = er[d] * 32.0f;                 // sqrt(1024) = 32
    float e2 = er[d + D_MODEL / 2] * 32.0f;
    orow[d]               = e1 * cs - e2 * sn;
    orow[d + D_MODEL / 2] = e2 * cs + e1 * sn;
  }
}

// a = silu(a) * b
__global__ void silu_mul(float* __restrict__ a, const float* __restrict__ b, size_t n)
{
  size_t i = (size_t)blockIdx.x * 256 + threadIdx.x;
  if (i < n) { float x = a[i]; a[i] = (x / (1.0f + __expf(-x))) * b[i]; }
}

// ---------------------------------------------------------------------------
// Host orchestration. Input order = jax pytree flatten (dict keys sorted):
// dec[6]{cattn{kb,kw,ob,ow,qb,qw,vb,vw}, ffn{b1,b2,b3,w1,w2,w3}, g_cattn,
//        g_ffn, g_sattn, ln1{b,g}, ln2{b,g}, ln3{b,g}, sattn{...}}  (31 ea)
// emb_src, emb_tgt,
// enc[6]{attn{...}, ffn{...}, g_attn, g_ffn, ln1{b,g}, ln2{b,g}}    (20 ea)
// out_b, out_w, src_tokens, tgt_tokens.
// ---------------------------------------------------------------------------
extern "C" void kernel_launch(void* const* d_in, const int* in_sizes, int n_in,
                              void* d_out, int out_size, void* d_ws, size_t ws_size,
                              hipStream_t stream)
{
  (void)in_sizes; (void)n_in; (void)out_size; (void)ws_size;
  auto F = [&](int i) { return (const float*)d_in[i]; };
  const int EMB_SRC = 186, EMB_TGT = 187, ENC0 = 188, OUT_B = 308, OUT_W = 309,
            SRC_TOK = 310, TGT_TOK = 311;

  float* ws = (float*)d_ws;
  size_t off = 0;
  float* src = ws + off; off += (size_t)B_SZ * S_SRC_L * D_MODEL;   // enc stream
  float* tgt = ws + off; off += (size_t)B_SZ * S_TGT_L * D_MODEL;   // dec stream
  float* xb  = ws + off; off += (size_t)B_SZ * S_SRC_L * D_MODEL;   // LN out
  float* qb  = ws + off; off += (size_t)B_SZ * S_SRC_L * D_MODEL;
  float* kb  = ws + off; off += (size_t)B_SZ * S_SRC_L * DHEAD;
  float* vb  = ws + off; off += (size_t)B_SZ * S_SRC_L * DHEAD;
  float* ab  = ws + off; off += (size_t)B_SZ * S_SRC_L * D_MODEL;   // attn out
  float* f1  = ws + off; off += (size_t)B_SZ * S_SRC_L * FFDIM;
  float* f2  = ws + off; off += (size_t)B_SZ * S_SRC_L * FFDIM;

  const int MSRC = B_SZ * S_SRC_L;   // 2048
  const int MTGT = B_SZ * S_TGT_L;   // 1024
  const float qscale = 0.125f;       // DH^-0.5

  auto gemm = [&](const float* A, const float* W, const float* bias,
                  const float* resid, const float* aptr, float cs,
                  float* C, int M, int N, int K) {
    dim3 g((N + 127) / 128, M / 128);
    gemm_bf16w<<<g, 256, 0, stream>>>(A, W, bias, resid, aptr, cs, C, M, N, K);
  };
  auto attn = [&](const float* Q, const float* Km, const float* Vm, float* O,
                  int T, int S, int causal) {
    mqa_flash<<<B_SZ * NHEAD * (T / 64), 128, 0, stream>>>(Q, Km, Vm, O, T, S, causal);
  };
  auto ln = [&](const float* X, const float* g_, const float* b_, float* Y, int rows) {
    ln_kernel<<<rows, 256, 0, stream>>>(X, g_, b_, Y, D_MODEL);
  };
  auto ffn = [&](float* stream_buf, int M, const float* b1, const float* b2,
                 const float* b3, const float* w1, const float* w2, const float* w3,
                 const float* gff) {
    gemm(xb, w1, b1, nullptr, nullptr, 1.0f, f1, M, FFDIM, D_MODEL);
    gemm(xb, w2, b2, nullptr, nullptr, 1.0f, f2, M, FFDIM, D_MODEL);
    size_t nel = (size_t)M * FFDIM;
    silu_mul<<<(unsigned)((nel + 255) / 256), 256, 0, stream>>>(f1, f2, nel);
    gemm(f1, w3, b3, stream_buf, gff, 1.0f, stream_buf, M, D_MODEL, FFDIM);
  };

  // Embeddings + RoPE
  embed_rope<<<MSRC, 256, 0, stream>>>((const int*)d_in[SRC_TOK], F(EMB_SRC), src, S_SRC_L);
  embed_rope<<<MTGT, 256, 0, stream>>>((const int*)d_in[TGT_TOK], F(EMB_TGT), tgt, S_TGT_L);

  // ---- Encoder ----
  for (int L = 0; L < NENC; ++L) {
    int p = ENC0 + L * 20;
    const float *a_kb=F(p+0), *a_kw=F(p+1), *a_ob=F(p+2), *a_ow=F(p+3),
                *a_qb=F(p+4), *a_qw=F(p+5), *a_vb=F(p+6), *a_vw=F(p+7);
    const float *f_b1=F(p+8), *f_b2=F(p+9), *f_b3=F(p+10),
                *f_w1=F(p+11), *f_w2=F(p+12), *f_w3=F(p+13);
    const float *g_at=F(p+14), *g_ff=F(p+15);
    const float *l1b=F(p+16), *l1g=F(p+17), *l2b=F(p+18), *l2g=F(p+19);

    ln(src, l1g, l1b, xb, MSRC);
    gemm(xb, a_qw, a_qb, nullptr, nullptr, qscale, qb, MSRC, D_MODEL, D_MODEL);
    gemm(xb, a_kw, a_kb, nullptr, nullptr, 1.0f, kb, MSRC, DHEAD, D_MODEL);
    gemm(xb, a_vw, a_vb, nullptr, nullptr, 1.0f, vb, MSRC, DHEAD, D_MODEL);
    attn(qb, kb, vb, ab, S_SRC_L, S_SRC_L, 0);
    gemm(ab, a_ow, a_ob, src, g_at, 1.0f, src, MSRC, D_MODEL, D_MODEL);
    ln(src, l2g, l2b, xb, MSRC);
    ffn(src, MSRC, f_b1, f_b2, f_b3, f_w1, f_w2, f_w3, g_ff);
  }

  // ---- Decoder ----
  for (int L = 0; L < NDEC; ++L) {
    int p = L * 31;
    const float *c_kb=F(p+0), *c_kw=F(p+1), *c_ob=F(p+2), *c_ow=F(p+3),
                *c_qb=F(p+4), *c_qw=F(p+5), *c_vb=F(p+6), *c_vw=F(p+7);
    const float *f_b1=F(p+8), *f_b2=F(p+9), *f_b3=F(p+10),
                *f_w1=F(p+11), *f_w2=F(p+12), *f_w3=F(p+13);
    const float *g_ca=F(p+14), *g_ff=F(p+15), *g_sa=F(p+16);
    const float *l1b=F(p+17), *l1g=F(p+18), *l2b=F(p+19), *l2g=F(p+20),
                *l3b=F(p+21), *l3g=F(p+22);
    const float *s_kb=F(p+23), *s_kw=F(p+24), *s_ob=F(p+25), *s_ow=F(p+26),
                *s_qb=F(p+27), *s_qw=F(p+28), *s_vb=F(p+29), *s_vw=F(p+30);

    // causal self-attention
    ln(tgt, l1g, l1b, xb, MTGT);
    gemm(xb, s_qw, s_qb, nullptr, nullptr, qscale, qb, MTGT, D_MODEL, D_MODEL);
    gemm(xb, s_kw, s_kb, nullptr, nullptr, 1.0f, kb, MTGT, DHEAD, D_MODEL);
    gemm(xb, s_vw, s_vb, nullptr, nullptr, 1.0f, vb, MTGT, DHEAD, D_MODEL);
    attn(qb, kb, vb, ab, S_TGT_L, S_TGT_L, 1);
    gemm(ab, s_ow, s_ob, tgt, g_sa, 1.0f, tgt, MTGT, D_MODEL, D_MODEL);
    // cross-attention (K/V from encoder memory, un-normed)
    ln(tgt, l2g, l2b, xb, MTGT);
    gemm(xb, c_qw, c_qb, nullptr, nullptr, qscale, qb, MTGT, D_MODEL, D_MODEL);
    gemm(src, c_kw, c_kb, nullptr, nullptr, 1.0f, kb, MSRC, DHEAD, D_MODEL);
    gemm(src, c_vw, c_vb, nullptr, nullptr, 1.0f, vb, MSRC, DHEAD, D_MODEL);
    attn(qb, kb, vb, ab, S_TGT_L, S_SRC_L, 0);
    gemm(ab, c_ow, c_ob, tgt, g_ca, 1.0f, tgt, MTGT, D_MODEL, D_MODEL);
    // FFN
    ln(tgt, l3g, l3b, xb, MTGT);
    ffn(tgt, MTGT, f_b1, f_b2, f_b3, f_w1, f_w2, f_w3, g_ff);
  }

  // ---- Output projection -> d_out [B*S_TGT, V_TGT] fp32 ----
  gemm(tgt, F(OUT_W), F(OUT_B), nullptr, nullptr, 1.0f,
       (float*)d_out, MTGT, VTGT, D_MODEL);
}
